// SSA_22771916603472
// MI455X (gfx1250) — matrix operand order repaired and checked by reading, not compiled, not run on hardware
//
#include <hip/hip_runtime.h>
#include <hip/hip_bf16.h>

// ---------------------------------------------------------------------------
// Problem constants (from reference): x [16, 256, 128, 128] fp32
// tokens T = 64 + 256 + 1024 + 4096 = 5440 (concat order k=16,8,4,2)
// ---------------------------------------------------------------------------
#define B_   16
#define C_   256
#define HW_  16384
#define T_   5440

typedef __attribute__((ext_vector_type(16))) __bf16 v16bf;
typedef __attribute__((ext_vector_type(8)))  float  v8f;

struct U32x2 { uint4 lo, hi; };   // 32 bytes == one v16bf fragment

static __device__ __forceinline__ v16bf frag_cast(uint4 lo, uint4 hi) {
    U32x2 s; s.lo = lo; s.hi = hi;
    return __builtin_bit_cast(v16bf, s);
}

// fp32 -> bf16 round-to-nearest-even (bit trick)
static __device__ __forceinline__ unsigned short f2bf(float f) {
    unsigned int u = __builtin_bit_cast(unsigned int, f);
    u += 0x7FFFu + ((u >> 16) & 1u);
    return (unsigned short)(u >> 16);
}

// ---------------------------------------------------------------------------
// Kernel 1: multiscale pooling -> bf16 tokens [B*C, 5440]
// One block per (b,c) image. Hierarchical 2x2 means through LDS
// (mean-of-means over equal counts == k x k mean).
// ---------------------------------------------------------------------------
__global__ __launch_bounds__(256)
void pool_tokens_kernel(const float* __restrict__ x, unsigned short* __restrict__ tok) {
    __shared__ float lds[5376];        // pool2[4096] | pool4[1024]@4096 | pool8[256]@5120
    const int img = blockIdx.x;        // b*C + c
    const int tid = threadIdx.x;
    const float2* src2 = (const float2*)(x + (size_t)img * HW_);
    unsigned short* trow = tok + (size_t)img * T_;

    // pool2: 64x64  -> tokens @1344
#pragma unroll
    for (int i = 0; i < 16; ++i) {
        int o  = tid + i * 256;
        int ro = o >> 6, co = o & 63;
        float2 a = src2[(2 * ro) * 64 + co];
        float2 b = src2[(2 * ro + 1) * 64 + co];
        float v = (a.x + a.y + b.x + b.y) * 0.25f;
        lds[o] = v;
        trow[1344 + o] = f2bf(v);
    }
    __syncthreads();

    // pool4: 32x32 -> tokens @320
#pragma unroll
    for (int i = 0; i < 4; ++i) {
        int o  = tid + i * 256;
        int ro = o >> 5, co = o & 31;
        float v = (lds[(2 * ro) * 64 + 2 * co]     + lds[(2 * ro) * 64 + 2 * co + 1] +
                   lds[(2 * ro + 1) * 64 + 2 * co] + lds[(2 * ro + 1) * 64 + 2 * co + 1]) * 0.25f;
        lds[4096 + o] = v;
        trow[320 + o] = f2bf(v);
    }
    __syncthreads();

    // pool8: 16x16 -> tokens @64
    {
        int o  = tid;
        int ro = o >> 4, co = o & 15;
        float v = (lds[4096 + (2 * ro) * 32 + 2 * co]     + lds[4096 + (2 * ro) * 32 + 2 * co + 1] +
                   lds[4096 + (2 * ro + 1) * 32 + 2 * co] + lds[4096 + (2 * ro + 1) * 32 + 2 * co + 1]) * 0.25f;
        lds[5120 + o] = v;
        trow[64 + o] = f2bf(v);
    }
    __syncthreads();

    // pool16: 8x8 -> tokens @0
    if (tid < 64) {
        int ro = tid >> 3, co = tid & 7;
        float v = (lds[5120 + (2 * ro) * 16 + 2 * co]     + lds[5120 + (2 * ro) * 16 + 2 * co + 1] +
                   lds[5120 + (2 * ro + 1) * 16 + 2 * co] + lds[5120 + (2 * ro + 1) * 16 + 2 * co + 1]) * 0.25f;
        trow[tid] = f2bf(v);
    }
}

// ---------------------------------------------------------------------------
// Kernel 2: E[b] = Tok · Tokᵀ  (256x256, K=5440) via v_wmma_f32_16x16x32_bf16.
// One wave per 16x16 output tile; tokens (44.5 MB) are L2-resident.
// A-frag per lane (row M=lane%16): K runs {k0+half*8..+7, k0+16+half*8..+7}.
// B-frag per lane (col N=lane%16): K runs {k0+half*8..+7, k0+16+half*8..+7}.
// For E = Tok·Tokᵀ, B[k][n] = Tok[n][k] -> B-frag is a contiguous row read too.
// ---------------------------------------------------------------------------
__global__ __launch_bounds__(256)
void energy_kernel(const unsigned short* __restrict__ tok, float* __restrict__ energy) {
    const int b    = blockIdx.x >> 5;
    const int wg   = blockIdx.x & 31;
    const int wave = threadIdx.x >> 5;
    const int lane = threadIdx.x & 31;
    const int half = lane >> 4;
    const int nl   = lane & 15;
    const int tile = wg * 8 + wave;    // 0..255
    const int ti   = tile >> 4;
    const int tj   = tile & 15;

    const unsigned short* arow = tok + (size_t)(b * C_ + ti * 16 + nl) * T_;
    const unsigned short* brow = tok + (size_t)(b * C_ + tj * 16 + nl) * T_;

    v8f acc = {};
    for (int k0 = 0; k0 < T_; k0 += 32) {
        uint4 alo = *(const uint4*)(arow + k0 + half * 8);
        uint4 ahi = *(const uint4*)(arow + k0 + 16 + half * 8);
        uint4 blo = *(const uint4*)(brow + k0 + half * 8);
        uint4 bhi = *(const uint4*)(brow + k0 + 16 + half * 8);
        acc = __builtin_amdgcn_wmma_f32_16x16x32_bf16(
                false, frag_cast(alo, ahi), false, frag_cast(blo, bhi),
                (short)0, acc, false, false);
    }

    float* Eb = energy + (size_t)b * (C_ * C_);
#pragma unroll
    for (int r = 0; r < 8; ++r) {
        int row = ti * 16 + half * 8 + r;      // C/D layout: VGPR r -> M = r + half*8
        int col = tj * 16 + nl;
        Eb[row * C_ + col] = acc[r];
    }
}

// ---------------------------------------------------------------------------
// Kernel 3: attention = softmax(max-E) = softmax(-E) per row -> bf16.
// Stable form: exp(minE - E) / sum. One block per row of 256.
// ---------------------------------------------------------------------------
__global__ __launch_bounds__(256)
void softmax_kernel(const float* __restrict__ energy, unsigned short* __restrict__ att) {
    __shared__ float red[256];
    const int row = blockIdx.x;
    const int tid = threadIdx.x;
    float e = energy[(size_t)row * C_ + tid];

    red[tid] = e;
    __syncthreads();
    for (int s = 128; s > 0; s >>= 1) {
        if (tid < s) red[tid] = fminf(red[tid], red[tid + s]);
        __syncthreads();
    }
    float mn = red[0];
    __syncthreads();

    float p = __expf(mn - e);
    red[tid] = p;
    __syncthreads();
    for (int s = 128; s > 0; s >>= 1) {
        if (tid < s) red[tid] += red[tid + s];
        __syncthreads();
    }
    float inv = 1.0f / red[0];
    att[(size_t)row * C_ + tid] = f2bf(p * inv);
}

// ---------------------------------------------------------------------------
// Kernel 4: out = gamma * (att · V) + x, V = x reshaped [C, HW].
// One WG per (b, 64-wide N strip); WG covers all M=256 rows so V is read once.
// x tile (32 K-rows x 64 N-cols) staged fp32->bf16 TRANSPOSED into LDS with
// row stride 40 halfwords (80 B): 16B-aligned b128 reads, conflict-free
// (20-dword stride, gcd trick -> all 64 banks hit once per 16 lanes).
// Each wave owns M rows [32w, 32w+32) x N 64 -> 8 accumulator tiles.
// ---------------------------------------------------------------------------
__global__ __launch_bounds__(256)
void av_kernel(const unsigned short* __restrict__ att,
               const float* __restrict__ x,
               const float* __restrict__ gammap,
               float* __restrict__ out) {
    __shared__ unsigned short ldsT[64 * 40];     // [col][k], padded stride 40
    const int b    = blockIdx.x >> 8;            // 256 strips per batch
    const int n0   = (blockIdx.x & 255) * 64;
    const int tid  = threadIdx.x;
    const int wave = tid >> 5;
    const int lane = tid & 31;
    const int half = lane >> 4;
    const int nl   = lane & 15;
    const float gamma = gammap[0];

    const float* xb           = x   + (size_t)b * (C_ * (size_t)HW_);
    const unsigned short* ab  = att + (size_t)b * (C_ * C_);

    v8f acc[2][4] = {};

    for (int kk = 0; kk < 8; ++kk) {
        const int k0 = kk * 32;
        // ---- stage: x[b, k0..k0+31, n0..n0+63] -> ldsT[col][k] (bf16) ----
        {
            int k = tid >> 3;                      // 0..31
            int j = tid & 7;                       // 0..7 -> cols j*8..+7
            const float* g = xb + (size_t)(k0 + k) * HW_ + n0 + j * 8;
            float4 f0 = *(const float4*)(g);
            float4 f1 = *(const float4*)(g + 4);
            int cb = j * 8;
            ldsT[(cb + 0) * 40 + k] = f2bf(f0.x);
            ldsT[(cb + 1) * 40 + k] = f2bf(f0.y);
            ldsT[(cb + 2) * 40 + k] = f2bf(f0.z);
            ldsT[(cb + 3) * 40 + k] = f2bf(f0.w);
            ldsT[(cb + 4) * 40 + k] = f2bf(f1.x);
            ldsT[(cb + 5) * 40 + k] = f2bf(f1.y);
            ldsT[(cb + 6) * 40 + k] = f2bf(f1.z);
            ldsT[(cb + 7) * 40 + k] = f2bf(f1.w);
        }
        __syncthreads();

        // ---- A fragments from attention (L2 resident) ----
        const unsigned short* ar0 = ab + (size_t)(wave * 32 + nl) * C_ + k0;
        const unsigned short* ar1 = ab + (size_t)(wave * 32 + 16 + nl) * C_ + k0;
        v16bf af0 = frag_cast(*(const uint4*)(ar0 + half * 8),
                              *(const uint4*)(ar0 + 16 + half * 8));
        v16bf af1 = frag_cast(*(const uint4*)(ar1 + half * 8),
                              *(const uint4*)(ar1 + 16 + half * 8));

        // ---- B fragments from LDS, 8 WMMAs per wave per k-step ----
#pragma unroll
        for (int nt = 0; nt < 4; ++nt) {
            int col = nt * 16 + nl;
            const unsigned short* lp = &ldsT[col * 40 + half * 8];
            v16bf bf = frag_cast(*(const uint4*)(lp), *(const uint4*)(lp + 16));
            acc[0][nt] = __builtin_amdgcn_wmma_f32_16x16x32_bf16(
                           false, af0, false, bf, (short)0, acc[0][nt], false, false);
            acc[1][nt] = __builtin_amdgcn_wmma_f32_16x16x32_bf16(
                           false, af1, false, bf, (short)0, acc[1][nt], false, false);
        }
        __syncthreads();
    }

    // ---- epilogue: gamma*acc + x (x re-read hits L2) ----
    float* ob = out + (size_t)b * (C_ * (size_t)HW_);
#pragma unroll
    for (int mt = 0; mt < 2; ++mt) {
#pragma unroll
        for (int nt = 0; nt < 4; ++nt) {
#pragma unroll
            for (int r = 0; r < 8; ++r) {
                int m = wave * 32 + mt * 16 + half * 8 + r;
                int n = n0 + nt * 16 + nl;
                size_t idx = (size_t)m * HW_ + n;
                ob[idx] = gamma * acc[mt][nt][r] + xb[idx];
            }
        }
    }
}

// ---------------------------------------------------------------------------
// Launch: ws layout = tokens bf16 (44,564,480 B) | energy f32 (4 MB) | att bf16 (2 MB)
// ---------------------------------------------------------------------------
extern "C" void kernel_launch(void* const* d_in, const int* in_sizes, int n_in,
                              void* d_out, int out_size, void* d_ws, size_t ws_size,
                              hipStream_t stream) {
    (void)in_sizes; (void)n_in; (void)out_size; (void)ws_size;
    const float* x     = (const float*)d_in[0];
    const float* gamma = (const float*)d_in[1];
    float* out         = (float*)d_out;

    const size_t tok_bytes = (size_t)B_ * C_ * T_ * 2;          // 44,564,480
    const size_t eng_bytes = (size_t)B_ * C_ * C_ * 4;          // 4,194,304
    unsigned short* tok  = (unsigned short*)d_ws;
    float*          eng  = (float*)((char*)d_ws + tok_bytes);
    unsigned short* att  = (unsigned short*)((char*)d_ws + tok_bytes + eng_bytes);

    pool_tokens_kernel<<<B_ * C_, 256, 0, stream>>>(x, tok);
    energy_kernel<<<B_ * 32, 256, 0, stream>>>(tok, eng);
    softmax_kernel<<<B_ * C_, 256, 0, stream>>>(eng, att);
    av_kernel<<<B_ * 256, 256, 0, stream>>>(att, x, gamma, out);
}
